// GraphConvBlock_3100966387879
// MI455X (gfx1250) — matrix operand order repaired and checked by reading, not compiled, not run on hardware
//
#include <hip/hip_runtime.h>
#include <hip/hip_bf16.h>

// ---------------------------------------------------------------------------
// GCN 2-layer block for MI455X (gfx1250): double-buffered TDM-staged bf16
// WMMA GEMMs + f32 atomic edge scatter.
//   layer: y = A0@W0 + (agg/deg)@W1 + b0 + b1 (+ReLU)
// ---------------------------------------------------------------------------

typedef __attribute__((ext_vector_type(16))) __bf16        v16bf;
typedef __attribute__((ext_vector_type(8)))  float         v8f;
typedef __attribute__((ext_vector_type(4)))  unsigned int  v4u;
typedef __attribute__((ext_vector_type(8)))  int           v8i;
typedef __attribute__((ext_vector_type(4)))  int           v4i;

__device__ __forceinline__ unsigned short f2bf(float f) {
  unsigned u = __float_as_uint(f);
  return (unsigned short)((u + 0x7FFFu + ((u >> 16) & 1u)) >> 16);
}

// ---- utility kernels -------------------------------------------------------

__global__ void zero_f32(float* __restrict__ p, long long n) {
  long long i = (long long)blockIdx.x * blockDim.x + threadIdx.x;
  long long s = (long long)gridDim.x * blockDim.x;
  for (; i < n; i += s) p[i] = 0.0f;
}

__global__ void deg_kernel(const int* __restrict__ edges, int E,
                           float* __restrict__ deg) {
  int i = blockIdx.x * blockDim.x + threadIdx.x;
  int s = gridDim.x * blockDim.x;
  for (; i < E; i += s) {
    atomicAdd(&deg[edges[2 * i + 0]], 1.0f);
    atomicAdd(&deg[edges[2 * i + 1]], 1.0f);
  }
}

__global__ void recip_kernel(float* __restrict__ deg, int n) {
  int i = blockIdx.x * blockDim.x + threadIdx.x;
  if (i < n) deg[i] = 1.0f / fmaxf(deg[i], 1.0f);
}

// f32 -> bf16 bulk convert
__global__ void cvt_bf16_kernel(const float* __restrict__ x,
                                unsigned short* __restrict__ y, long long n) {
  long long i = (long long)blockIdx.x * blockDim.x + threadIdx.x;
  long long s = (long long)gridDim.x * blockDim.x;
  for (; i < n; i += s) y[i] = f2bf(x[i]);
}

// aggb[i] = bf16(agg[i] * invdeg[i >> shift])   (shift = log2(F))
__global__ void finalize_agg_kernel(const float* __restrict__ agg,
                                    const float* __restrict__ invd,
                                    unsigned short* __restrict__ aggb,
                                    int shift, long long n) {
  long long i = (long long)blockIdx.x * blockDim.x + threadIdx.x;
  long long s = (long long)gridDim.x * blockDim.x;
  for (; i < n; i += s) aggb[i] = f2bf(agg[i] * invd[i >> shift]);
}

// W[k,F] f32  ->  Wt[f,K] bf16 (transposed, k contiguous)
__global__ void wt_bf16_kernel(const float* __restrict__ W,
                               unsigned short* __restrict__ Wt, int K, int F) {
  int i = blockIdx.x * blockDim.x + threadIdx.x;  // over F*K
  if (i < K * F) {
    int f = i / K, k = i - f * K;
    Wt[i] = f2bf(W[(size_t)k * F + f]);
  }
}

// One wave per (edge, direction); lanes stride the feature row.
__global__ void scatter_kernel(const int* __restrict__ edges, long long ndir,
                               const float* __restrict__ x,
                               float* __restrict__ agg, int F) {
  const int lane = threadIdx.x & 31;
  long long w  = (long long)blockIdx.x * (blockDim.x >> 5) + (threadIdx.x >> 5);
  long long nw = (long long)gridDim.x * (blockDim.x >> 5);
  for (; w < ndir; w += nw) {
    long long e = w >> 1;
    int dir = (int)(w & 1);
    int s = edges[2 * e + dir];        // src
    int d = edges[2 * e + (dir ^ 1)];  // dst
    const float* xs = x + (size_t)s * F;
    float*       ad = agg + (size_t)d * F;
    // prefetch next task's source row (global_prefetch_b8)
    long long wn = w + nw;
    if (wn < ndir) {
      long long en = wn >> 1;
      int sn = edges[2 * en + (int)(wn & 1)];
      __builtin_prefetch(x + (size_t)sn * F + lane, 0, 1);
    }
    for (int f = lane; f < F; f += 32) atomicAdd(&ad[f], xs[f]);
  }
}

// ---- Tensor Data Mover staging ---------------------------------------------
// 2D descriptor: rows x 32 bf16 tile, row stride in elements. Packs D# group0
// (count/lds_addr/global_addr/type) + group1 (data_size, dims, tile, stride).
// Toolchain uses the 6-arg builtin: (v4u, v8i, v4i, v4i, v8i, i32 cpol).
__device__ __forceinline__ void tdm_load_2d(unsigned lds_off,
                                            const unsigned short* gptr,
                                            unsigned rows,
                                            unsigned long long stride_elems) {
  unsigned long long ga = (unsigned long long)(size_t)gptr;
  v4u g0;
  g0[0] = 1u;                                            // count = 1 (valid)
  g0[1] = lds_off;                                       // lds_addr
  g0[2] = (unsigned)ga;                                  // global_addr lo
  g0[3] = (unsigned)((ga >> 32) & 0x1FFFFFFu) | (2u << 30);  // addr hi | type=2
  const unsigned dim0 = 32u, tile0 = 32u;
  v8i g1;
  g1[0] = 0x00010000;                                    // data_size = 2 bytes
  g1[1] = (int)((dim0 & 0xFFFFu) << 16);                 // tensor_dim0[15:0]
  g1[2] = (int)((dim0 >> 16) | ((rows & 0xFFFFu) << 16));// dim0 hi | dim1 lo
  g1[3] = (int)((rows >> 16) | (tile0 << 16));           // dim1 hi | tile_dim0
  g1[4] = (int)(rows & 0xFFFFu);                         // tile_dim1 (tile2=0)
  g1[5] = (int)(unsigned)(stride_elems & 0xFFFFFFFFull); // dim0_stride lo32
  g1[6] = (int)(unsigned)((stride_elems >> 32) & 0xFFFFull); // stride hi16
  g1[7] = 0;
  v4i gz4 = {0, 0, 0, 0};
  v8i gz8 = {0, 0, 0, 0, 0, 0, 0, 0};
  __builtin_amdgcn_tensor_load_to_lds(g0, g1, gz4, gz4, gz8, 0);
}

// ---- fused WMMA GEMM (double-buffered TDM pipeline) ------------------------
// out = A0@W0t' + A1@W1t' + b0 + b1, block tile 128(M) x 64(N), 8 waves.
__global__ __launch_bounds__(256)
void gcn_gemm_kernel(const unsigned short* __restrict__ A0,
                     const unsigned short* __restrict__ A1,
                     const unsigned short* __restrict__ B0t,   // [F,K] bf16
                     const unsigned short* __restrict__ B1t,   // [F,K] bf16
                     const float* __restrict__ bias0,
                     const float* __restrict__ bias1,
                     float* __restrict__ outf,
                     unsigned short* __restrict__ outb,        // may be null
                     int M, int K, int F, int relu) {
  __shared__ unsigned short As0[2][128 * 32];
  __shared__ unsigned short As1[2][128 * 32];
  __shared__ unsigned short Bs0[2][64 * 32];
  __shared__ unsigned short Bs1[2][64 * 32];

  const int t    = threadIdx.x;
  const int lane = t & 31;
  const int wave = t >> 5;
  const int half = lane >> 4;
  const int n    = lane & 15;
  const int base = blockIdx.x * 128;
  const int fb   = blockIdx.y * 64;

  int rows = M - base;
  if (rows > 128) rows = 128;

  auto issue_tiles = [&](int buf, int kk) {
    tdm_load_2d((unsigned)(size_t)As0[buf], A0 + (size_t)base * K + kk,
                (unsigned)rows, (unsigned long long)K);
    tdm_load_2d((unsigned)(size_t)As1[buf], A1 + (size_t)base * K + kk,
                (unsigned)rows, (unsigned long long)K);
    tdm_load_2d((unsigned)(size_t)Bs0[buf], B0t + (size_t)fb * K + kk, 64u,
                (unsigned long long)K);
    tdm_load_2d((unsigned)(size_t)Bs1[buf], B1t + (size_t)fb * K + kk, 64u,
                (unsigned long long)K);
  };

  v8f acc[4] = {{}, {}, {}, {}};

  const int steps = K >> 5;  // K / 32
  if (wave == 0) issue_tiles(0, 0);  // prologue: fill buffer 0

  for (int it = 0; it < steps; ++it) {
    const int cur = it & 1;
    if (wave == 0) __builtin_amdgcn_s_wait_tensorcnt(0);  // buf[cur] landed
    __syncthreads();  // release: buf[cur] ready; buf[cur^1] fully consumed
    if (wave == 0 && it + 1 < steps) issue_tiles(cur ^ 1, (it + 1) << 5);

    union Frag { v16bf v; unsigned u[8]; };
    Frag a0, a1;
    const unsigned* Ar0 =
        (const unsigned*)As0[cur] + (size_t)(wave * 16 + n) * 16;
    const unsigned* Ar1 =
        (const unsigned*)As1[cur] + (size_t)(wave * 16 + n) * 16;
#pragma unroll
    for (int p = 0; p < 8; ++p) {
      // A (16x32 bf16): elem e -> k = 16*half + e
      int ai = 8 * half + p;
      a0.u[p] = Ar0[ai];
      a1.u[p] = Ar1[ai];
    }
#pragma unroll
    for (int j = 0; j < 4; ++j) {
      const unsigned* Br0 =
          (const unsigned*)Bs0[cur] + (size_t)(j * 16 + n) * 16 + 8 * half;
      const unsigned* Br1 =
          (const unsigned*)Bs1[cur] + (size_t)(j * 16 + n) * 16 + 8 * half;
      Frag bf0, bf1;
#pragma unroll
      for (int p = 0; p < 8; ++p) {
        // B (32x16 bf16): elem e -> k = 16*half + e (k contiguous per column)
        bf0.u[p] = Br0[p];
        bf1.u[p] = Br1[p];
      }
      acc[j] = __builtin_amdgcn_wmma_f32_16x16x32_bf16(
          false, a0.v, false, bf0.v, (short)0, acc[j], false, false);
      acc[j] = __builtin_amdgcn_wmma_f32_16x16x32_bf16(
          false, a1.v, false, bf1.v, (short)0, acc[j], false, false);
    }
    __syncthreads();  // everyone done with buf[cur] before it is re-filled
  }

  const int tile_m = blockIdx.x * 8 + wave;
  if (tile_m * 16 < M) {  // M % 16 == 0 -> tiles are all-in or all-out
#pragma unroll
    for (int j = 0; j < 4; ++j) {
      const int col = fb + j * 16 + n;
      const float bias = bias0[col] + bias1[col];
#pragma unroll
      for (int r = 0; r < 8; ++r) {
        int row = tile_m * 16 + 8 * half + r;
        float v = acc[j][r] + bias;
        if (relu) v = fmaxf(v, 0.0f);
        outf[(size_t)row * F + col] = v;
        if (outb) outb[(size_t)row * F + col] = f2bf(v);
      }
    }
  }
}

// ---------------------------------------------------------------------------

extern "C" void kernel_launch(void* const* d_in, const int* in_sizes, int n_in,
                              void* d_out, int out_size, void* d_ws,
                              size_t ws_size, hipStream_t stream) {
  const float* x    = (const float*)d_in[0];
  const int*   edg  = (const int*)d_in[1];
  const float* W0_0 = (const float*)d_in[2];
  const float* b0_0 = (const float*)d_in[3];
  const float* W1_0 = (const float*)d_in[4];
  const float* b1_0 = (const float*)d_in[5];
  const float* W0_1 = (const float*)d_in[6];
  const float* b0_1 = (const float*)d_in[7];
  const float* W1_1 = (const float*)d_in[8];
  const float* b1_1 = (const float*)d_in[9];
  float* out = (float*)d_out;

  const int F_IN = 256, F_HID = 512, F_OUT = 256;
  const int N = in_sizes[0] / F_IN;  // 50000
  const int E = in_sizes[1] / 2;     // 400000
  const long long NDIR = 2LL * E;

  // workspace layout
  char* w = (char*)d_ws;
  size_t o = 0;
  const size_t NP = ((size_t)N + 63) & ~(size_t)63;
  float* deg  = (float*)(w + o); o += NP * 4;
  float* agg0 = (float*)(w + o); o += (size_t)N * F_IN * 4;
  float* agg1 = (float*)(w + o); o += (size_t)N * F_HID * 4;
  float* h    = (float*)(w + o); o += (size_t)N * F_HID * 4;
  unsigned short* xb    = (unsigned short*)(w + o); o += (size_t)N * F_IN * 2;
  unsigned short* aggb0 = (unsigned short*)(w + o); o += (size_t)N * F_IN * 2;
  unsigned short* hb    = (unsigned short*)(w + o); o += (size_t)N * F_HID * 2;
  unsigned short* aggb1 = (unsigned short*)(w + o); o += (size_t)N * F_HID * 2;
  unsigned short* Wt00  = (unsigned short*)(w + o); o += (size_t)F_IN * F_HID * 2;
  unsigned short* Wt10  = (unsigned short*)(w + o); o += (size_t)F_IN * F_HID * 2;
  unsigned short* Wt01  = (unsigned short*)(w + o); o += (size_t)F_HID * F_OUT * 2;
  unsigned short* Wt11  = (unsigned short*)(w + o); o += (size_t)F_HID * F_OUT * 2;

  // 1) zero deg + agg0 + agg1 (contiguous)
  long long nz = (long long)NP + (long long)N * (F_IN + F_HID);
  zero_f32<<<2048, 256, 0, stream>>>(deg, nz);

  // 2) degree -> 1/max(deg,1)
  deg_kernel<<<1024, 256, 0, stream>>>(edg, E, deg);
  recip_kernel<<<(N + 255) / 256, 256, 0, stream>>>(deg, N);

  // 3) one-time bf16 conversions: x and transposed weights
  cvt_bf16_kernel<<<2048, 256, 0, stream>>>(x, xb, (long long)N * F_IN);
  wt_bf16_kernel<<<(F_IN * F_HID + 255) / 256, 256, 0, stream>>>(W0_0, Wt00,
                                                                 F_IN, F_HID);
  wt_bf16_kernel<<<(F_IN * F_HID + 255) / 256, 256, 0, stream>>>(W1_0, Wt10,
                                                                 F_IN, F_HID);
  wt_bf16_kernel<<<(F_HID * F_OUT + 255) / 256, 256, 0, stream>>>(W0_1, Wt01,
                                                                  F_HID, F_OUT);
  wt_bf16_kernel<<<(F_HID * F_OUT + 255) / 256, 256, 0, stream>>>(W1_1, Wt11,
                                                                  F_HID, F_OUT);

  // 4) layer-0 aggregation + scale/convert
  scatter_kernel<<<4096, 256, 0, stream>>>(edg, NDIR, x, agg0, F_IN);
  finalize_agg_kernel<<<2048, 256, 0, stream>>>(agg0, deg, aggb0, 8,
                                                (long long)N * F_IN);

  // 5) layer 0: h = relu(x@W0 + agg0n@W1 + b)  (f32 + bf16 copies)
  const int mtiles = (N + 15) / 16;
  dim3 g0((mtiles + 7) / 8, F_HID / 64);
  gcn_gemm_kernel<<<g0, 256, 0, stream>>>(xb, aggb0, Wt00, Wt10, b0_0, b1_0, h,
                                          hb, N, F_IN, F_HID, 1);

  // 6) layer-1 aggregation + scale/convert
  scatter_kernel<<<4096, 256, 0, stream>>>(edg, NDIR, h, agg1, F_HID);
  finalize_agg_kernel<<<2048, 256, 0, stream>>>(agg1, deg, aggb1, 9,
                                                (long long)N * F_HID);

  // 7) layer 1: out = h@W0 + agg1n@W1 + b
  dim3 g1((mtiles + 7) / 8, F_OUT / 64);
  gcn_gemm_kernel<<<g1, 256, 0, stream>>>(hb, aggb1, Wt01, Wt11, b0_1, b1_1,
                                          out, nullptr, N, F_HID, F_OUT, 0);
}
